// NeighborhoodAttention_7842610282588
// MI455X (gfx1250) — compile-verified, matrix operand order checked by reading, not run on hardware
//
#include <hip/hip_runtime.h>
#include <stdint.h>

// ---------------------------------------------------------------------------
// Types for CDNA5 WMMA (wave32).  v_wmma_f32_16x16x32_bf16:
//   A: 16x32 bf16, B: 32x16 bf16, C/D: 16x16 f32.
// Per-lane fragment model (ISA 7.12.2): lane L holds row/col (L&15); L>=16
// selects the second K-half; the 16 bf16 are two contiguous 8-element runs at
// K offsets kh*8 and kh*8+16.  C/D: vgpr r -> M = r + 8*(L>=16), N = L&15.
// ---------------------------------------------------------------------------
typedef __attribute__((ext_vector_type(4)))  unsigned int u32x4;
typedef __attribute__((ext_vector_type(16))) __bf16       v16bf;
typedef __attribute__((ext_vector_type(8)))  float        v8f;

union Frag { u32x4 u[2]; v16bf v; };

__device__ __forceinline__ unsigned short f2bf(float f) {
  unsigned int u = __float_as_uint(f);
  u += 0x7FFFu + ((u >> 16) & 1u);          // round-to-nearest-even
  return (unsigned short)(u >> 16);
}
__device__ __forceinline__ float bf2f(unsigned short h) {
  return __uint_as_float(((unsigned int)h) << 16);
}

#define B_      4
#define HW_     56
#define DIMK    512
#define HEADS_  16
#define HD      32          // head dim
#define BHW     12544       // 4*56*56
#define PIX     3136        // 56*56
#define NTOT    1536        // 3*512 fused QKV columns
#define BSTRIDE 264         // LDS B-tile stride: 528B -> 4-bank step, 16B aligned

// ---------------------------------------------------------------------------
// Kernel 1: f32 -> bf16 convert (memory bound, trivial)
// ---------------------------------------------------------------------------
__global__ void cvt_f32_bf16(const float* __restrict__ src,
                             unsigned short* __restrict__ dst, int n) {
  int i = blockIdx.x * blockDim.x + threadIdx.x;
  if (i < n) dst[i] = f2bf(src[i]);
}

// ---------------------------------------------------------------------------
// Kernel 2: fused QKV projection GEMM  C[12544,1536] = Xbf @ Wbf^T
//   Block: 256 thr = 8 waves; wave w -> rows [Mbase + 16w, +16), 64 cols.
//   B tile (weight rows == WMMA-B K-contiguous lanes) staged in LDS in two
//   256-K phases.  All 4 B fragments live in distinct registers so the 4
//   WMMAs issue back-to-back after one dscnt wait; the A global load is
//   pipelined one k-step ahead.  Epilogue: +bias, Q * 1/sqrt(32), scatter
//   to [b,head,i,j,d] bf16.
// ---------------------------------------------------------------------------
__global__ void __launch_bounds__(256)
qkv_gemm(const unsigned short* __restrict__ X,   // [12544][512] bf16
         const unsigned short* __restrict__ W,   // [1536][512]  bf16
         const float* __restrict__ bq, const float* __restrict__ bk,
         const float* __restrict__ bv,
         unsigned short* __restrict__ QKV) {     // 3 x [b,head,56,56,32] bf16
  __shared__ unsigned short Bl[64 * BSTRIDE];    // 33 KB

  const int tid  = threadIdx.x;
  const int wave = tid >> 5, lane = tid & 31;
  const int m16  = lane & 15, kh = lane >> 4;
  const int Nbase = blockIdx.x * 64;
  const int Mbase = blockIdx.y * 128 + wave * 16;

  v8f acc[4];
  #pragma unroll
  for (int t = 0; t < 4; ++t) acc[t] = (v8f){};

  const unsigned short* arow = X + (size_t)(Mbase + m16) * DIMK + kh * 8;

  for (int phase = 0; phase < 2; ++phase) {
    __syncthreads();
    // stage 64 cols x 256 K of W into LDS (16B chunks); prefetch next phase
    for (int idx = tid; idx < 64 * 32; idx += 256) {
      int col = idx >> 5, ch = idx & 31;
      const unsigned short* wp = W + (size_t)(Nbase + col) * DIMK + phase * 256 + ch * 8;
      u32x4 d = *(const u32x4*)wp;
      if (phase == 0) __builtin_prefetch(wp + 256, 0, 0);  // global_prefetch_b8
      *(u32x4*)&Bl[col * BSTRIDE + ch * 8] = d;
    }
    __syncthreads();

    // A fragment for ks=0, pipelined one step ahead thereafter
    Frag a;
    a.u[0] = *(const u32x4*)(arow + phase * 256);
    a.u[1] = *(const u32x4*)(arow + phase * 256 + 16);

    for (int ks = 0; ks < 8; ++ks) {
      const int k = phase * 256 + ks * 32;
      // all 4 B fragments in distinct registers -> one wait, 4 WMMAs
      Frag bfr[4];
      #pragma unroll
      for (int t = 0; t < 4; ++t) {
        const unsigned short* bp = &Bl[(t * 16 + m16) * BSTRIDE + ks * 32 + kh * 8];
        bfr[t].u[0] = *(const u32x4*)bp;
        bfr[t].u[1] = *(const u32x4*)(bp + 16);
      }
      Frag an = a;
      if (ks < 7) {
        an.u[0] = *(const u32x4*)(arow + k + 32);
        an.u[1] = *(const u32x4*)(arow + k + 48);
      }
      #pragma unroll
      for (int t = 0; t < 4; ++t)
        acc[t] = __builtin_amdgcn_wmma_f32_16x16x32_bf16(
            false, a.v, false, bfr[t].v, (short)0, acc[t], false, false);
      a = an;
    }
  }

  // epilogue: bias + (Q only) 1/sqrt(32) scale, scatter to [b,head,i,j,d]
  #pragma unroll
  for (int t = 0; t < 4; ++t) {
    const int jj   = Nbase + t * 16 + m16;
    const int mi   = jj >> 9;             // 0=Q 1=K 2=V
    const int j    = jj & 511;
    const float* bp = (mi == 0) ? bq : (mi == 1) ? bk : bv;
    const float bias  = bp[j];
    const float scale = (mi == 0) ? 0.17677669529663687f : 1.0f;
    const int head = j >> 5, dd = j & 31;
    unsigned short* ob = QKV + (size_t)mi * ((size_t)BHW * DIMK);
    #pragma unroll
    for (int r = 0; r < 8; ++r) {
      int row = Mbase + r + (kh << 3);
      int b   = row / PIX;
      int rem = row - b * PIX;            // = i*56 + w
      size_t dst = ((size_t)((b * HEADS_ + head) * PIX + rem)) * HD + dd;
      ob[dst] = f2bf((acc[t][r] + bias) * scale);
    }
  }
}

// ---------------------------------------------------------------------------
// Kernel 3: neighborhood attention, one WG per (b, head, 8x8 query tile).
//   Union of neighborhoods = 14x14 key patch (196 keys, pad to 224).
//   S = Q K^T   : 4 qtiles x 14 ktiles single-k-step WMMAs across 8 waves
//                 (all 7 B fragments preloaded -> WMMA burst)
//   softmax     : threads 0..63 (2 full waves), masked 49-entry rows + rpb,
//                 P written in place of S (rows are thread-private)
//   ctx = P V   : 7 accumulating WMMAs per wave, fragments preloaded, f32 out
// ---------------------------------------------------------------------------
#define KSTR 40     // K-tile LDS stride (80B: conflict-free, 16B aligned)
#define PSTR 232    // S/P and Vt stride (464B: conflict-free, 16B aligned)

__global__ void __launch_bounds__(256)
natten(const unsigned short* __restrict__ Q,
       const unsigned short* __restrict__ K,
       const unsigned short* __restrict__ V,
       const float* __restrict__ rpb,      // [16][13][13] f32
       float* __restrict__ out) {          // [4][56][56][512] f32
  __shared__ unsigned short Kl[224 * KSTR];   // 17.5 KB  [key][d]
  __shared__ unsigned short Vt[32  * PSTR];   // 14.5 KB  [d][key] (transposed)
  __shared__ unsigned short SP[64  * PSTR];   // 29 KB    scores, then probs

  const int tid  = threadIdx.x;
  const int tile = blockIdx.x, head = blockIdx.y, b = blockIdx.z;
  const int i0 = (tile / 7) * 8, j0 = (tile % 7) * 8;
  int rb = i0 - 3; rb = rb < 0 ? 0 : (rb > 42 ? 42 : rb);
  int cb = j0 - 3; cb = cb < 0 ? 0 : (cb > 42 ? 42 : cb);
  const size_t base = (size_t)(b * HEADS_ + head) * PIX * HD;

  // ---- stage K (row-major) and V (transposed); zero-pad keys 196..223 ----
  for (int idx = tid; idx < 224 * 4; idx += 256) {
    int pos = idx >> 2, ch = idx & 3;
    u32x4 kd = {0, 0, 0, 0}, vd = {0, 0, 0, 0};
    if (pos < 196) {
      int r = rb + pos / 14, c = cb + pos % 14;
      size_t off = base + (size_t)(r * HW_ + c) * HD + ch * 8;
      kd = *(const u32x4*)(K + off);
      vd = *(const u32x4*)(V + off);
    }
    *(u32x4*)&Kl[pos * KSTR + ch * 8] = kd;
    const unsigned short* vs = (const unsigned short*)&vd;
    #pragma unroll
    for (int e = 0; e < 8; ++e) Vt[(ch * 8 + e) * PSTR + pos] = vs[e];
  }
  __syncthreads();

  const int wave = tid >> 5, lane = tid & 31;
  const int m16  = lane & 15, kh = lane >> 4;
  const int qt   = wave >> 1;

  // ---- S = Q K^T (d = 32 = exactly one WMMA K-step) ----
  {
    const int kg = wave & 1;                 // key-tile group: 7 tiles each
    const int q  = qt * 16 + m16;
    const int qi = i0 + (q >> 3), qj = j0 + (q & 7);
    const unsigned short* qp = Q + base + (size_t)(qi * HW_ + qj) * HD + kh * 8;
    Frag a;
    a.u[0] = *(const u32x4*)qp;
    a.u[1] = *(const u32x4*)(qp + 16);
    // preload all 7 key-tile B fragments (distinct regs -> WMMA burst)
    Frag bfr[7];
    #pragma unroll
    for (int t = 0; t < 7; ++t) {
      const unsigned short* kp = &Kl[((kg * 7 + t) * 16 + m16) * KSTR + kh * 8];
      bfr[t].u[0] = *(const u32x4*)kp;
      bfr[t].u[1] = *(const u32x4*)(kp + 16);
    }
    #pragma unroll
    for (int t = 0; t < 7; ++t) {
      v8f s = (v8f){};
      s = __builtin_amdgcn_wmma_f32_16x16x32_bf16(
          false, a.v, false, bfr[t].v, (short)0, s, false, false);
      const int qrow = qt * 16 + (kh << 3);
      const int col  = (kg * 7 + t) * 16 + m16;
      #pragma unroll
      for (int r = 0; r < 8; ++r) SP[(qrow + r) * PSTR + col] = f2bf(s[r]);
    }
  }
  __syncthreads();

  // ---- masked softmax with rpb; rows thread-private -> S reused as P ----
  if (tid < 64) {                           // waves 0,1 fully active
    const int q = tid;
    const int i = i0 + (q >> 3), j = j0 + (q & 7);
    int ni = i - 3; ni = ni < 0 ? 0 : (ni > 49 ? 49 : ni);
    int nj = j - 3; nj = nj < 0 ? 0 : (nj > 49 ? 49 : nj);
    float sc[49];
    float mx = -1e30f;
    #pragma unroll
    for (int ki = 0; ki < 7; ++ki) {
      const int r = ni + ki, bi = r - i + 6;
      #pragma unroll
      for (int kj = 0; kj < 7; ++kj) {
        const int c = nj + kj, bj = c - j + 6;
        const int pos = (r - rb) * 14 + (c - cb);
        float v = bf2f(SP[q * PSTR + pos]) + rpb[(head * 13 + bi) * 13 + bj];
        sc[ki * 7 + kj] = v;
        mx = fmaxf(mx, v);
      }
    }
    float sum = 0.f;
    #pragma unroll
    for (int t = 0; t < 49; ++t) { sc[t] = __expf(sc[t] - mx); sum += sc[t]; }
    const float inv = 1.f / sum;
    for (int t = 0; t < PSTR; ++t) SP[q * PSTR + t] = 0;   // zero row (all reads done)
    #pragma unroll
    for (int ki = 0; ki < 7; ++ki) {
      const int r = ni + ki;
      #pragma unroll
      for (int kj = 0; kj < 7; ++kj) {
        const int c = nj + kj;
        const int pos = (r - rb) * 14 + (c - cb);
        SP[q * PSTR + pos] = f2bf(sc[ki * 7 + kj] * inv);
      }
    }
  }
  __syncthreads();

  // ---- ctx = P V : 224 keys = 7 accumulating WMMA K-steps ----
  {
    const int nt = wave & 1;                // which 16 of the 32 head dims
    const int q  = qt * 16 + m16;
    // preload all 7 (A,B) fragment pairs -> accumulation WMMAs back-to-back
    Frag a[7], bfr[7];
    #pragma unroll
    for (int ks = 0; ks < 7; ++ks) {
      const unsigned short* pp = &SP[q * PSTR + ks * 32 + kh * 8];
      a[ks].u[0] = *(const u32x4*)pp;
      a[ks].u[1] = *(const u32x4*)(pp + 16);
      const unsigned short* vp = &Vt[(nt * 16 + m16) * PSTR + ks * 32 + kh * 8];
      bfr[ks].u[0] = *(const u32x4*)vp;
      bfr[ks].u[1] = *(const u32x4*)(vp + 16);
    }
    v8f acc = (v8f){};
    #pragma unroll
    for (int ks = 0; ks < 7; ++ks)
      acc = __builtin_amdgcn_wmma_f32_16x16x32_bf16(
          false, a[ks].v, false, bfr[ks].v, (short)0, acc, false, false);

    const int dcol = head * HD + nt * 16 + m16;
    #pragma unroll
    for (int r = 0; r < 8; ++r) {
      const int qq = qt * 16 + r + (kh << 3);
      const int i = i0 + (qq >> 3), j = j0 + (qq & 7);
      out[(size_t)(b * PIX + i * HW_ + j) * DIMK + dcol] = acc[r];
    }
  }
}

// ---------------------------------------------------------------------------
// Host launch: convert -> fused QKV GEMM -> neighborhood attention
// ---------------------------------------------------------------------------
extern "C" void kernel_launch(void* const* d_in, const int* in_sizes, int n_in,
                              void* d_out, int out_size, void* d_ws, size_t ws_size,
                              hipStream_t stream) {
  (void)in_sizes; (void)n_in; (void)out_size; (void)ws_size;
  const float* hidden = (const float*)d_in[0];
  const float* Wq = (const float*)d_in[1];
  const float* bq = (const float*)d_in[2];
  const float* Wk = (const float*)d_in[3];
  const float* bk = (const float*)d_in[4];
  const float* Wv = (const float*)d_in[5];
  const float* bv = (const float*)d_in[6];
  const float* rp = (const float*)d_in[7];
  float* out = (float*)d_out;

  unsigned short* ws = (unsigned short*)d_ws;
  const size_t NX = (size_t)BHW * DIMK;           // 6,422,528 elems
  const size_t NW = (size_t)DIMK * DIMK;          //   262,144 elems
  unsigned short* Xb = ws;                        // bf16 hidden
  unsigned short* Wb = Xb + NX;                   // bf16 packed [Wq;Wk;Wv]
  unsigned short* Qb = Wb + 3 * NW;               // bf16 Q (pre-scaled)
  unsigned short* Kb = Qb + NX;
  unsigned short* Vb = Kb + NX;                   // total ~53 MB of ws

  cvt_f32_bf16<<<(unsigned)((NX + 255) / 256), 256, 0, stream>>>(hidden, Xb, (int)NX);
  cvt_f32_bf16<<<(unsigned)((NW + 255) / 256), 256, 0, stream>>>(Wq, Wb,          (int)NW);
  cvt_f32_bf16<<<(unsigned)((NW + 255) / 256), 256, 0, stream>>>(Wk, Wb + NW,     (int)NW);
  cvt_f32_bf16<<<(unsigned)((NW + 255) / 256), 256, 0, stream>>>(Wv, Wb + 2 * NW, (int)NW);

  qkv_gemm<<<dim3(NTOT / 64, BHW / 128), 256, 0, stream>>>(Xb, Wb, bq, bk, bv, Qb);

  natten<<<dim3(49, HEADS_, B_), 256, 0, stream>>>(Qb, Kb, Vb, rp, out);
}
